// MHLA_Normed_Torch_42752104464795
// MI455X (gfx1250) — matrix operand order repaired and checked by reading, not compiled, not run on hardware
//
#include <hip/hip_runtime.h>
#include <hip/hip_bf16.h>

// ---------------- types ----------------
typedef __bf16 bf16x8  __attribute__((ext_vector_type(8)));
typedef __bf16 bf16x16 __attribute__((ext_vector_type(16)));
typedef float  floatx8 __attribute__((ext_vector_type(8)));

static __device__ __forceinline__ __bf16 to_bf16(float f) { return (__bf16)f; }

static __device__ __forceinline__ floatx8 zero8() {
    floatx8 z;
#pragma unroll
    for (int i = 0; i < 8; ++i) z[i] = 0.0f;
    return z;
}

static __device__ __forceinline__ bf16x16 cat16(bf16x8 lo, bf16x8 hi) {
    bf16x16 r;
#pragma unroll
    for (int i = 0; i < 8; ++i) { r[i] = lo[i]; r[8 + i] = hi[i]; }
    return r;
}

static __device__ __forceinline__ floatx8 wmma_bf16(bf16x16 a, bf16x16 b, floatx8 c) {
    // D = A(16x32) * B(32x16) + C, fp32 accumulate, gfx1250 wave32
    return __builtin_amdgcn_wmma_f32_16x16x32_bf16(false, a, false, b, (short)0, c, false, false);
}

// ---------------- problem constants ----------------
#define NB   8
#define NBLK 256
#define WTOK 49
#define CH   512
#define NH   8
#define DH   64
#define IMGS 112
#define TOKS 100352ull          // NB*NBLK*WTOK
#define BHN  16384              // NB*NH*NBLK
#define EPSF 1e-6f

// ---------------- workspace layout ----------------
static constexpr size_t AL(size_t x) { return (x + 255) & ~(size_t)255; }
static constexpr size_t O_XN   = 0;                                  // bf16 TOKS*CH
static constexpr size_t O_WQKV = AL(O_XN   + TOKS * 512 * 2);        // bf16 1536*512
static constexpr size_t O_WOUT = AL(O_WQKV + 1536ull * 512 * 2);     // bf16 512*512
static constexpr size_t O_WMF  = AL(O_WOUT + 512ull * 512 * 2);      // f32  256*256
static constexpr size_t O_WMB  = AL(O_WMF  + 256ull * 256 * 4);      // bf16 256*256
static constexpr size_t O_Q    = AL(O_WMB  + 256ull * 256 * 2);      // bf16 [B,H,N,W,D]
static constexpr size_t O_K    = AL(O_Q    + TOKS * 512 * 2);        // bf16 [B,H,N,W,D]
static constexpr size_t O_VF   = AL(O_K    + TOKS * 512 * 2);        // f32  [B,N,W,C]
static constexpr size_t O_VB   = AL(O_VF   + TOKS * 512 * 4);        // bf16 [B,H,N,W,D]
static constexpr size_t O_KV   = AL(O_VB   + TOKS * 512 * 2);        // bf16 [BHN,64,64]
static constexpr size_t O_KVM  = AL(O_KV   + (size_t)BHN * 4096 * 2);// bf16 [BHN,64,64]
static constexpr size_t O_KS   = AL(O_KVM  + (size_t)BHN * 4096 * 2);// f32  [BHN,64]
static constexpr size_t O_QKS  = AL(O_KS   + (size_t)BHN * 64 * 4);  // f32  [BHN,49]
static constexpr size_t O_NRM  = AL(O_QKS  + (size_t)BHN * 49 * 4);  // f32  [BHN,49]
static constexpr size_t O_ATT  = AL(O_NRM  + (size_t)BHN * 49 * 4);  // f32  [B,N,W,C]
static constexpr size_t O_CMB  = AL(O_ATT  + TOKS * 512 * 4);        // bf16 [B,N,W,C]

// =====================================================================
// 0a. Convert w_qkv / w_out to bf16
// =====================================================================
__global__ void k_cvt_weights(const float* __restrict__ wqkv, const float* __restrict__ wout,
                              __bf16* __restrict__ wqkvb, __bf16* __restrict__ woutb) {
    int i = blockIdx.x * 256 + threadIdx.x;
    if (i < 1536 * 512) wqkvb[i] = to_bf16(wqkv[i]);
    if (i < 512 * 512)  woutb[i] = to_bf16(wout[i]);
}

// =====================================================================
// 0b. _block_weights cos matrix (column-normalized), 16x16 grid
// =====================================================================
__global__ void k_weightmat(float* __restrict__ wf, __bf16* __restrict__ wb) {
    const int j = threadIdx.x;
    const float cr = (float)(j >> 4) + 0.5f;
    const float cc = (float)(j & 15) + 0.5f;
    const float inv_dmax = 1.0f / (sqrtf(2.0f) * 15.0f);
    const float pi4 = 0.78539816339744831f;
    float sum = 0.0f;
    for (int i = 0; i < 256; ++i) {
        float dr = (float)(i >> 4) + 0.5f - cr;
        float dc = (float)(i & 15) + 0.5f - cc;
        sum += cosf(sqrtf(dr * dr + dc * dc) * inv_dmax * pi4);
    }
    const float inv = 1.0f / sum;
    for (int i = 0; i < 256; ++i) {
        float dr = (float)(i >> 4) + 0.5f - cr;
        float dc = (float)(i & 15) + 0.5f - cc;
        float v = cosf(sqrtf(dr * dr + dc * dc) * inv_dmax * pi4) * inv;
        wf[i * 256 + j] = v;
        wb[i * 256 + j] = to_bf16(v);
    }
}

// =====================================================================
// 1. LayerNorm over C=512 -> bf16 (one 128-thread block per token)
// =====================================================================
__global__ __launch_bounds__(128) void k_layernorm(const float* __restrict__ x,
                                                   const float* __restrict__ g,
                                                   const float* __restrict__ bta,
                                                   __bf16* __restrict__ xn) {
    const int t = blockIdx.x;
    const int tid = threadIdx.x;
    const float4 v = ((const float4*)(x + (size_t)t * CH))[tid];
    float s  = v.x + v.y + v.z + v.w;
    float s2 = v.x * v.x + v.y * v.y + v.z * v.z + v.w * v.w;
    __shared__ float sh1[128], sh2[128];
    sh1[tid] = s; sh2[tid] = s2;
    __syncthreads();
    for (int o = 64; o > 0; o >>= 1) {
        if (tid < o) { sh1[tid] += sh1[tid + o]; sh2[tid] += sh2[tid + o]; }
        __syncthreads();
    }
    const float mean = sh1[0] * (1.0f / CH);
    const float var  = sh2[0] * (1.0f / CH) - mean * mean;
    const float rstd = rsqrtf(var + 1e-5f);
    const int c = tid * 4;
    float vv[4] = { v.x, v.y, v.z, v.w };
#pragma unroll
    for (int jj = 0; jj < 4; ++jj)
        xn[(size_t)t * CH + c + jj] = to_bf16((vv[jj] - mean) * rstd * g[c + jj] + bta[c + jj]);
}

// =====================================================================
// 2. qkv GEMM: [TOKS,512] x [1536,512]^T, WMMA bf16, 128x128x32 tiles.
//    Section (q/k/v) is uniform per block: sec = blockIdx.y >> 2.
//    Branchless prefetch of the next K-step tile (wraps on last iter).
// =====================================================================
__global__ __launch_bounds__(256) void k_gemm_qkv(const __bf16* __restrict__ A,
                                                  const __bf16* __restrict__ Wt,
                                                  __bf16* __restrict__ qb, __bf16* __restrict__ kb,
                                                  float* __restrict__ vf, __bf16* __restrict__ vb) {
    const int tid = threadIdx.x;
    const int Mbase = blockIdx.x * 128;
    const int Nbase = blockIdx.y * 128;
    __shared__ __align__(16) __bf16 sA[128 * 40];
    __shared__ __align__(16) __bf16 sB[128 * 40];
    const int lane = tid & 31, wv = tid >> 5;
    const int wm = wv & 1, wn = wv >> 1;
    const int lrow = lane & 15, hsel = lane >> 4;
    const int lr = tid >> 2, lc = (tid & 3) * 8;

    floatx8 acc[4][2];
#pragma unroll
    for (int i = 0; i < 4; ++i)
#pragma unroll
        for (int j = 0; j < 2; ++j) acc[i][j] = zero8();

    for (int kk = 0; kk < 512; kk += 32) {
        const int kp = (kk + 32) & 511;              // branchless next-tile prefetch offset
#pragma unroll
        for (int p = 0; p < 2; ++p) {
            int row = lr + p * 64;
            *(bf16x8*)&sA[row * 40 + lc] = *(const bf16x8*)&A[(size_t)(Mbase + row) * 512 + kk + lc];
            *(bf16x8*)&sB[row * 40 + lc] = *(const bf16x8*)&Wt[(size_t)(Nbase + row) * 512 + kk + lc];
            __builtin_prefetch(&A[(size_t)(Mbase + row) * 512 + kp + lc], 0, 3);
            __builtin_prefetch(&Wt[(size_t)(Nbase + row) * 512 + kp + lc], 0, 3);
        }
        __syncthreads();
        bf16x16 af[4], bfr[2];
#pragma unroll
        for (int i = 0; i < 4; ++i) {
            int r = wm * 64 + i * 16 + lrow;
            bf16x8 lo = *(const bf16x8*)&sA[r * 40 + hsel * 8];
            bf16x8 hi = *(const bf16x8*)&sA[r * 40 + 16 + hsel * 8];
            af[i] = cat16(lo, hi);
        }
#pragma unroll
        for (int j = 0; j < 2; ++j) {
            int r = wn * 32 + j * 16 + lrow;
            bf16x8 lo = *(const bf16x8*)&sB[r * 40 + hsel * 8];
            bf16x8 hi = *(const bf16x8*)&sB[r * 40 + 16 + hsel * 8];
            bfr[j] = cat16(lo, hi);
        }
#pragma unroll
        for (int i = 0; i < 4; ++i)
#pragma unroll
            for (int j = 0; j < 2; ++j) acc[i][j] = wmma_bf16(af[i], bfr[j], acc[i][j]);
        __syncthreads();
    }

    // block-uniform section: 0=q, 1=k, 2=v
    const int sec = blockIdx.y >> 2;
    const int colbase = (blockIdx.y & 3) * 128;
    if (sec == 0) {
#pragma unroll
        for (int i = 0; i < 4; ++i)
#pragma unroll
            for (int g = 0; g < 8; ++g) {
                const int mrow = Mbase + wm * 64 + i * 16 + hsel * 8 + g;
                const int b = mrow / (NBLK * WTOK);
                const int rem = mrow % (NBLK * WTOK);
                const int n = rem / WTOK, w = rem % WTOK;
#pragma unroll
                for (int j = 0; j < 2; ++j) {
                    const int cc = colbase + wn * 32 + j * 16 + lrow;
                    const int h = cc >> 6, d = cc & 63;
                    float val = fmaxf(acc[i][j][g], 0.0f) + EPSF;
                    qb[(((size_t)(b * NH + h) * NBLK + n) * WTOK + w) * DH + d] = to_bf16(val);
                }
            }
    } else if (sec == 1) {
#pragma unroll
        for (int i = 0; i < 4; ++i)
#pragma unroll
            for (int g = 0; g < 8; ++g) {
                const int mrow = Mbase + wm * 64 + i * 16 + hsel * 8 + g;
                const int b = mrow / (NBLK * WTOK);
                const int rem = mrow % (NBLK * WTOK);
                const int n = rem / WTOK, w = rem % WTOK;
#pragma unroll
                for (int j = 0; j < 2; ++j) {
                    const int cc = colbase + wn * 32 + j * 16 + lrow;
                    const int h = cc >> 6, d = cc & 63;
                    float val = fmaxf(acc[i][j][g], 0.0f) + EPSF;
                    kb[(((size_t)(b * NH + h) * NBLK + n) * WTOK + w) * DH + d] = to_bf16(val);
                }
            }
    } else {
#pragma unroll
        for (int i = 0; i < 4; ++i)
#pragma unroll
            for (int g = 0; g < 8; ++g) {
                const int mrow = Mbase + wm * 64 + i * 16 + hsel * 8 + g;
                const int b = mrow / (NBLK * WTOK);
                const int rem = mrow % (NBLK * WTOK);
                const int n = rem / WTOK, w = rem % WTOK;
#pragma unroll
                for (int j = 0; j < 2; ++j) {
                    const int cc = colbase + wn * 32 + j * 16 + lrow;
                    const int h = cc >> 6, d = cc & 63;
                    float val = acc[i][j][g];
                    vf[(size_t)mrow * CH + cc] = val;
                    vb[(((size_t)(b * NH + h) * NBLK + n) * WTOK + w) * DH + d] = to_bf16(val);
                }
            }
    }
}

// =====================================================================
// 3. kv = k^T v per (b,h,n) (64x64, K=49 padded to 64) + k_sum.
//    k and v stored TRANSPOSED into LDS -> all fragment loads b128.
// =====================================================================
__global__ __launch_bounds__(128) void k_kv(const __bf16* __restrict__ kb,
                                            const __bf16* __restrict__ vb,
                                            __bf16* __restrict__ kv, float* __restrict__ ksum) {
    const int bid = blockIdx.x;                      // (b*8+h)*256+n
    const int tid = threadIdx.x;
    __shared__ __align__(16) __bf16 skT[64 * 72];    // [d][w]
    __shared__ __align__(16) __bf16 svT[64 * 72];    // [e][w]
#pragma unroll
    for (int p = 0; p < 4; ++p) {
        int cidx = tid + p * 128;
        int row = cidx >> 3, col = (cidx & 7) * 8;   // row = w, col = d/e base
        bf16x8 kvv, vvv;
        if (row < WTOK) {
            kvv = *(const bf16x8*)&kb[((size_t)bid * WTOK + row) * DH + col];
            vvv = *(const bf16x8*)&vb[((size_t)bid * WTOK + row) * DH + col];
        } else {
#pragma unroll
            for (int e = 0; e < 8; ++e) { kvv[e] = (__bf16)0.0f; vvv[e] = (__bf16)0.0f; }
        }
#pragma unroll
        for (int e = 0; e < 8; ++e) {
            skT[(col + e) * 72 + row] = kvv[e];
            svT[(col + e) * 72 + row] = vvv[e];
        }
    }
    __syncthreads();
    if (tid < 64) {
        float s = 0.0f;
        for (int w = 0; w < WTOK; ++w) s += (float)skT[tid * 72 + w];
        ksum[(size_t)bid * 64 + tid] = s;
    }
    const int lane = tid & 31, wv = tid >> 5;
    const int lrow = lane & 15, hsel = lane >> 4;
    floatx8 acc[4];
#pragma unroll
    for (int j = 0; j < 4; ++j) acc[j] = zero8();
#pragma unroll
    for (int kk = 0; kk < 64; kk += 32) {
        const int dm = wv * 16 + lrow;               // A row (d)
        bf16x8 alo = *(const bf16x8*)&skT[dm * 72 + kk + hsel * 8];
        bf16x8 ahi = *(const bf16x8*)&skT[dm * 72 + kk + 16 + hsel * 8];
        bf16x16 af = cat16(alo, ahi);
#pragma unroll
        for (int j = 0; j < 4; ++j) {
            const int col = j * 16 + lrow;           // B col (e)
            bf16x8 blo = *(const bf16x8*)&svT[col * 72 + kk + hsel * 8];
            bf16x8 bhi = *(const bf16x8*)&svT[col * 72 + kk + 16 + hsel * 8];
            acc[j] = wmma_bf16(af, cat16(blo, bhi), acc[j]);
        }
    }
#pragma unroll
    for (int j = 0; j < 4; ++j)
#pragma unroll
        for (int g = 0; g < 8; ++g) {
            int d = wv * 16 + hsel * 8 + g;
            int e = j * 16 + lrow;
            kv[((size_t)bid * 64 + d) * 64 + e] = to_bf16(acc[j][g]);
        }
}

// =====================================================================
// 4. kvm = weight * kv (batched over 64 (b,h): M=256,K=256,N=4096).
//    B tile stored transposed in LDS -> contiguous b128 fragment loads.
// =====================================================================
__global__ __launch_bounds__(256) void k_mix(const __bf16* __restrict__ wmat,
                                             const __bf16* __restrict__ kv,
                                             __bf16* __restrict__ kvm) {
    const int bh = blockIdx.z;
    const int Mbase = blockIdx.y * 64;
    const int Nbase = blockIdx.x * 128;
    __shared__ __align__(16) __bf16 sA[64 * 40];     // [i][k]
    __shared__ __align__(16) __bf16 sBT[128 * 40];   // [n][k]
    const int tid = threadIdx.x, lane = tid & 31, wv = tid >> 5;
    const int wm = wv & 1, wn = wv >> 1, lrow = lane & 15, hsel = lane >> 4;
    const size_t kvbase = (size_t)bh * 256 * 4096;
    floatx8 acc[2][2];
#pragma unroll
    for (int i = 0; i < 2; ++i)
#pragma unroll
        for (int j = 0; j < 2; ++j) acc[i][j] = zero8();

    for (int kk = 0; kk < 256; kk += 32) {
        { int row = tid >> 2, col = (tid & 3) * 8;
          *(bf16x8*)&sA[row * 40 + col] =
              *(const bf16x8*)&wmat[(size_t)(Mbase + row) * 256 + kk + col]; }
#pragma unroll
        for (int p = 0; p < 2; ++p) {
            int cidx = tid + p * 256;
            int row = cidx >> 4, col = (cidx & 15) * 8;   // row = k, col = n base
            bf16x8 v = *(const bf16x8*)&kv[kvbase + (size_t)(kk + row) * 4096 + Nbase + col];
#pragma unroll
            for (int e = 0; e < 8; ++e) sBT[(col + e) * 40 + row] = v[e];
        }
        __syncthreads();
        bf16x16 af[2], bfr[2];
#pragma unroll
        for (int i = 0; i < 2; ++i) {
            int r = wm * 32 + i * 16 + lrow;
            bf16x8 lo = *(const bf16x8*)&sA[r * 40 + hsel * 8];
            bf16x8 hi = *(const bf16x8*)&sA[r * 40 + 16 + hsel * 8];
            af[i] = cat16(lo, hi);
        }
#pragma unroll
        for (int j = 0; j < 2; ++j) {
            const int col = wn * 32 + j * 16 + lrow;
            bf16x8 lo = *(const bf16x8*)&sBT[col * 40 + hsel * 8];
            bf16x8 hi = *(const bf16x8*)&sBT[col * 40 + 16 + hsel * 8];
            bfr[j] = cat16(lo, hi);
        }
#pragma unroll
        for (int i = 0; i < 2; ++i)
#pragma unroll
            for (int j = 0; j < 2; ++j) acc[i][j] = wmma_bf16(af[i], bfr[j], acc[i][j]);
        __syncthreads();
    }
#pragma unroll
    for (int i = 0; i < 2; ++i)
#pragma unroll
        for (int j = 0; j < 2; ++j)
#pragma unroll
            for (int g = 0; g < 8; ++g) {
                int irow = Mbase + wm * 32 + i * 16 + hsel * 8 + g;
                int ncol = Nbase + wn * 32 + j * 16 + lrow;
                kvm[kvbase + (size_t)irow * 4096 + ncol] = to_bf16(acc[i][j][g]);
            }
}

// =====================================================================
// 5a. qk_sum = q . k_sum
// =====================================================================
__global__ void k_qksum(const __bf16* __restrict__ qb, const float* __restrict__ ksum,
                        float* __restrict__ qks, int total) {
    int i = blockIdx.x * blockDim.x + threadIdx.x;
    if (i >= total) return;
    const int bid = i / WTOK, w = i % WTOK;
    const __bf16* qr = &qb[((size_t)bid * WTOK + w) * DH];
    const float* ks = &ksum[(size_t)bid * 64];
    float s = 0.0f;
#pragma unroll
    for (int d = 0; d < 64; ++d) s += (float)qr[d] * ks[d];
    qks[i] = s;
}

// =====================================================================
// 5b. normalizer = weight-mix of qk_sum + eps
// =====================================================================
__global__ void k_norm(const float* __restrict__ wmat, const float* __restrict__ qks,
                       float* __restrict__ nrm, int total) {
    int i = blockIdx.x * blockDim.x + threadIdx.x;
    if (i >= total) return;
    const int bh = i / (NBLK * WTOK);
    const int rem = i % (NBLK * WTOK);
    const int iblk = rem / WTOK, w = rem % WTOK;
    const float* wrow = &wmat[(size_t)iblk * 256];
    float s = 0.0f;
    for (int j = 0; j < 256; ++j) s += wrow[j] * qks[((size_t)bh * NBLK + j) * WTOK + w];
    nrm[i] = s + EPSF;
}

// =====================================================================
// 6. out = q @ kvm / normalizer -> attn [B,N,W,C] f32.
//    kvm tile stored transposed in LDS -> contiguous b128 fragment loads.
// =====================================================================
__global__ __launch_bounds__(128) void k_attnout(const __bf16* __restrict__ qb,
                                                 const __bf16* __restrict__ kvm,
                                                 const float* __restrict__ nrm,
                                                 float* __restrict__ attn) {
    const int bid = blockIdx.x;                      // (b*8+h)*256+n
    const int tid = threadIdx.x;
    __shared__ __align__(16) __bf16 sq[64 * 72];     // [w][d] row-major
    __shared__ __align__(16) __bf16 skvT[64 * 72];   // [e][d]
#pragma unroll
    for (int p = 0; p < 4; ++p) {
        int cidx = tid + p * 128;
        int row = cidx >> 3, col = (cidx & 7) * 8;
        bf16x8 qv;
        if (row < WTOK) qv = *(const bf16x8*)&qb[((size_t)bid * WTOK + row) * DH + col];
        else {
#pragma unroll
            for (int e = 0; e < 8; ++e) qv[e] = (__bf16)0.0f;
        }
        *(bf16x8*)&sq[row * 72 + col] = qv;
        bf16x8 kvv = *(const bf16x8*)&kvm[(size_t)bid * 4096 + row * 64 + col];  // row=d, col=e
#pragma unroll
        for (int e = 0; e < 8; ++e) skvT[(col + e) * 72 + row] = kvv[e];
    }
    __syncthreads();
    const int lane = tid & 31, wv = tid >> 5;
    const int lrow = lane & 15, hsel = lane >> 4;
    floatx8 acc[4];
#pragma unroll
    for (int j = 0; j < 4; ++j) acc[j] = zero8();
#pragma unroll
    for (int kk = 0; kk < 64; kk += 32) {
        const int r = wv * 16 + lrow;
        bf16x8 lo = *(const bf16x8*)&sq[r * 72 + kk + hsel * 8];
        bf16x8 hi = *(const bf16x8*)&sq[r * 72 + kk + 16 + hsel * 8];
        bf16x16 af = cat16(lo, hi);
#pragma unroll
        for (int j = 0; j < 4; ++j) {
            const int col = j * 16 + lrow;
            bf16x8 blo = *(const bf16x8*)&skvT[col * 72 + kk + hsel * 8];
            bf16x8 bhi = *(const bf16x8*)&skvT[col * 72 + kk + 16 + hsel * 8];
            acc[j] = wmma_bf16(af, cat16(blo, bhi), acc[j]);
        }
    }
    const int h = (bid >> 8) & 7, b = bid >> 11, n = bid & 255;
#pragma unroll
    for (int j = 0; j < 4; ++j)
#pragma unroll
        for (int g = 0; g < 8; ++g) {
            int w = wv * 16 + hsel * 8 + g;
            if (w < WTOK) {
                int e = j * 16 + lrow;
                float nv = nrm[(size_t)bid * WTOK + w];
                attn[((size_t)(b * NBLK + n) * WTOK + w) * CH + h * DH + e] = acc[j][g] / nv;
            }
        }
}

// =====================================================================
// 7. LePE depthwise 5x5 conv on 112x112 image + add attn -> combined bf16
// =====================================================================
__global__ void k_lepe(const float* __restrict__ vf, const float* __restrict__ lw,
                       const float* __restrict__ lb, const float* __restrict__ attn,
                       __bf16* __restrict__ comb) {
    size_t t = (size_t)blockIdx.x * blockDim.x + threadIdx.x;
    if (t >= TOKS * CH) return;
    const int c = (int)(t & 511);
    size_t rest = t >> 9;
    const int x = (int)(rest % IMGS); rest /= IMGS;
    const int y = (int)(rest % IMGS);
    const int b = (int)(rest / IMGS);
    float acc = lb[c];
#pragma unroll
    for (int dy = 0; dy < 5; ++dy) {
        int yy = y + dy - 2;
        if (yy < 0 || yy >= IMGS) continue;
#pragma unroll
        for (int dx = 0; dx < 5; ++dx) {
            int xx = x + dx - 2;
            if (xx < 0 || xx >= IMGS) continue;
            int n = (yy / 7) * 16 + (xx / 7);
            int w = (yy % 7) * 7 + (xx % 7);
            acc += vf[((size_t)(b * NBLK + n) * WTOK + w) * CH + c] * lw[c * 25 + dy * 5 + dx];
        }
    }
    const int n0 = (y / 7) * 16 + (x / 7);
    const int w0 = (y % 7) * 7 + (x % 7);
    const size_t oidx = ((size_t)(b * NBLK + n0) * WTOK + w0) * CH + c;
    comb[oidx] = to_bf16(attn[oidx] + acc);
}

// =====================================================================
// 8. Final GEMM: combined[TOKS,512] x w_out[512,512]^T + b_out -> d_out f32
// =====================================================================
__global__ __launch_bounds__(256) void k_gemm_out(const __bf16* __restrict__ A,
                                                  const __bf16* __restrict__ Wt,
                                                  const float* __restrict__ bias,
                                                  float* __restrict__ out) {
    const int tid = threadIdx.x;
    const int Mbase = blockIdx.x * 128;
    const int Nbase = blockIdx.y * 128;
    __shared__ __align__(16) __bf16 sA[128 * 40];
    __shared__ __align__(16) __bf16 sB[128 * 40];
    const int lane = tid & 31, wv = tid >> 5;
    const int wm = wv & 1, wn = wv >> 1;
    const int lrow = lane & 15, hsel = lane >> 4;
    const int lr = tid >> 2, lc = (tid & 3) * 8;
    floatx8 acc[4][2];
#pragma unroll
    for (int i = 0; i < 4; ++i)
#pragma unroll
        for (int j = 0; j < 2; ++j) acc[i][j] = zero8();

    for (int kk = 0; kk < 512; kk += 32) {
        const int kp = (kk + 32) & 511;              // branchless next-tile prefetch offset
#pragma unroll
        for (int p = 0; p < 2; ++p) {
            int row = lr + p * 64;
            *(bf16x8*)&sA[row * 40 + lc] = *(const bf16x8*)&A[(size_t)(Mbase + row) * 512 + kk + lc];
            *(bf16x8*)&sB[row * 40 + lc] = *(const bf16x8*)&Wt[(size_t)(Nbase + row) * 512 + kk + lc];
            __builtin_prefetch(&A[(size_t)(Mbase + row) * 512 + kp + lc], 0, 3);
        }
        __syncthreads();
        bf16x16 af[4], bfr[2];
#pragma unroll
        for (int i = 0; i < 4; ++i) {
            int r = wm * 64 + i * 16 + lrow;
            bf16x8 lo = *(const bf16x8*)&sA[r * 40 + hsel * 8];
            bf16x8 hi = *(const bf16x8*)&sA[r * 40 + 16 + hsel * 8];
            af[i] = cat16(lo, hi);
        }
#pragma unroll
        for (int j = 0; j < 2; ++j) {
            int r = wn * 32 + j * 16 + lrow;
            bf16x8 lo = *(const bf16x8*)&sB[r * 40 + hsel * 8];
            bf16x8 hi = *(const bf16x8*)&sB[r * 40 + 16 + hsel * 8];
            bfr[j] = cat16(lo, hi);
        }
#pragma unroll
        for (int i = 0; i < 4; ++i)
#pragma unroll
            for (int j = 0; j < 2; ++j) acc[i][j] = wmma_bf16(af[i], bfr[j], acc[i][j]);
        __syncthreads();
    }
#pragma unroll
    for (int i = 0; i < 4; ++i)
#pragma unroll
        for (int j = 0; j < 2; ++j)
#pragma unroll
            for (int g = 0; g < 8; ++g) {
                int mrow = Mbase + wm * 64 + i * 16 + hsel * 8 + g;
                int ocol = Nbase + wn * 32 + j * 16 + lrow;
                out[(size_t)mrow * 512 + ocol] = acc[i][j][g] + bias[ocol];
            }
}

// =====================================================================
// launch
// =====================================================================
extern "C" void kernel_launch(void* const* d_in, const int* in_sizes, int n_in,
                              void* d_out, int out_size, void* d_ws, size_t ws_size,
                              hipStream_t stream) {
    (void)in_sizes; (void)n_in; (void)out_size; (void)ws_size;
    const float* x      = (const float*)d_in[0];
    const float* ln_g   = (const float*)d_in[1];
    const float* ln_b   = (const float*)d_in[2];
    const float* w_qkv  = (const float*)d_in[3];
    const float* lepe_w = (const float*)d_in[4];
    const float* lepe_b = (const float*)d_in[5];
    const float* w_out  = (const float*)d_in[6];
    const float* b_out  = (const float*)d_in[7];
    float* out = (float*)d_out;

    char* ws = (char*)d_ws;
    __bf16* xn    = (__bf16*)(ws + O_XN);
    __bf16* wqkvb = (__bf16*)(ws + O_WQKV);
    __bf16* woutb = (__bf16*)(ws + O_WOUT);
    float*  wmf   = (float*)(ws + O_WMF);
    __bf16* wmb   = (__bf16*)(ws + O_WMB);
    __bf16* qb    = (__bf16*)(ws + O_Q);
    __bf16* kb    = (__bf16*)(ws + O_K);
    float*  vf    = (float*)(ws + O_VF);
    __bf16* vb    = (__bf16*)(ws + O_VB);
    __bf16* kv    = (__bf16*)(ws + O_KV);
    __bf16* kvm   = (__bf16*)(ws + O_KVM);
    float*  ksum  = (float*)(ws + O_KS);
    float*  qks   = (float*)(ws + O_QKS);
    float*  nrm   = (float*)(ws + O_NRM);
    float*  attn  = (float*)(ws + O_ATT);
    __bf16* comb  = (__bf16*)(ws + O_CMB);

    k_cvt_weights<<<3072, 256, 0, stream>>>(w_qkv, w_out, wqkvb, woutb);
    k_weightmat<<<1, 256, 0, stream>>>(wmf, wmb);
    k_layernorm<<<(int)TOKS, 128, 0, stream>>>(x, ln_g, ln_b, xn);
    k_gemm_qkv<<<dim3(784, 12), 256, 0, stream>>>(xn, wqkvb, qb, kb, vf, vb);
    k_kv<<<BHN, 128, 0, stream>>>(kb, vb, kv, ksum);
    k_mix<<<dim3(32, 4, 64), 256, 0, stream>>>(wmb, kv, kvm);
    {
        int total = BHN * WTOK;
        k_qksum<<<(total + 255) / 256, 256, 0, stream>>>(qb, ksum, qks, total);
        k_norm<<<(total + 255) / 256, 256, 0, stream>>>(wmf, qks, nrm, total);
    }
    k_attnout<<<BHN, 128, 0, stream>>>(qb, kvm, nrm, attn);
    {
        size_t total = TOKS * CH;
        k_lepe<<<(unsigned)((total + 255) / 256), 256, 0, stream>>>(vf, lepe_w, lepe_b, attn, comb);
    }
    k_gemm_out<<<dim3(784, 4), 256, 0, stream>>>(comb, woutb, b_out, out);
}
